// NeuraLogicHelperLayer_87316685128562
// MI455X (gfx1250) — matrix operand order repaired and checked by reading, not compiled, not run on hardware
//
#include <hip/hip_runtime.h>

#define FD 128  // feature dim D (fixed by the reference)

typedef __bf16 bf16_t;
typedef __attribute__((ext_vector_type(16))) __bf16 v16bf;
typedef __attribute__((ext_vector_type(8)))  float  v8f;

// ---------- helpers ----------
static __device__ __forceinline__ unsigned pack_bf16_rne(float a, float b) {
    unsigned ua = __builtin_bit_cast(unsigned, a);
    unsigned ub = __builtin_bit_cast(unsigned, b);
    ua += 0x7fffu + ((ua >> 16) & 1u);   // round-to-nearest-even into bf16
    ub += 0x7fffu + ((ub >> 16) & 1u);
    return (ua >> 16) | (ub & 0xffff0000u);
}

// ---------- 1) zero the S accumulator (bf16) ----------
__global__ __launch_bounds__(256) void zero_kernel(uint4* __restrict__ p, long n16) {
    long i = (long)blockIdx.x * blockDim.x + threadIdx.x;
    if (i < n16) p[i] = uint4{0u, 0u, 0u, 0u};
}

// ---------- 2) convert W fp32 -> bf16 ----------
__global__ __launch_bounds__(256) void convw_kernel(const float* __restrict__ W,
                                                    unsigned* __restrict__ Wb,
                                                    long npairs) {
    long i = (long)blockIdx.x * blockDim.x + threadIdx.x;
    if (i < npairs) Wb[i] = pack_bf16_rne(W[2 * i], W[2 * i + 1]);
}

// ---------- 3) edge scatter: S[w, v, :] += x[u, :]  (packed bf16 atomics) ----------
__global__ __launch_bounds__(256) void scatter_kernel(const float* __restrict__ x,
                                                      const int* __restrict__ u,
                                                      const int* __restrict__ v,
                                                      const int* __restrict__ widx,
                                                      bf16_t* __restrict__ Sb,
                                                      int N, long E) {
    const int lane = threadIdx.x & 31;
    const int wave = threadIdx.x >> 5;
    const long e   = (long)blockIdx.x * 8 + wave;   // one wave per edge
    if (e >= E) return;

    const int su = u[e];
    const int dv = v[e];
    const int w  = widx[e];

    const float4 xv = *(const float4*)(x + (size_t)su * FD + lane * 4);
    const unsigned p0 = pack_bf16_rne(xv.x, xv.y);
    const unsigned p1 = pack_bf16_rne(xv.z, xv.w);

    bf16_t* dst = Sb + ((size_t)w * (size_t)N + (size_t)dv) * FD + lane * 4;
    unsigned long long a0 = (unsigned long long)dst;
    unsigned long long a1 = (unsigned long long)(dst + 2);

    // CDNA5 packed-bf16 atomic add, no return (STOREcnt-tracked)
    asm volatile("global_atomic_pk_add_bf16 %0, %1, off" :: "v"(a0), "v"(p0) : "memory");
    asm volatile("global_atomic_pk_add_bf16 %0, %1, off" :: "v"(a1), "v"(p1) : "memory");
}

// ---------- 4) grouped GEMM + tanh:  out[v,i] = tanh( sum_w sum_j W[w,i,j] * S[w,v,j] ) ----------
// Block = 256 threads = 8 waves. Block tile: 128 rows x 128 cols.
// Wave tile: 16 rows x 128 cols (8 accumulators of 16x16), K = NW*128 in steps of 32.
__global__ __launch_bounds__(256) void gemm_tanh_kernel(const bf16_t* __restrict__ Sb,
                                                        const bf16_t* __restrict__ Wb,
                                                        float* __restrict__ out,
                                                        int N, int NW) {
    const int lane = threadIdx.x & 31;
    const int wave = threadIdx.x >> 5;
    const int c    = lane & 15;   // A: row-in-tile, B: col-in-tile, C: col-in-tile
    const int half = lane >> 4;   // K half selector

    const long rowBase = (long)blockIdx.x * 128 + wave * 16;
    long arow = rowBase + c;
    if (arow >= N) arow = N - 1;                 // clamp (stores are guarded)

    v8f acc[8];
#pragma unroll
    for (int n = 0; n < 8; ++n) acc[n] = (v8f)(0.0f);

    const int kbMax = NW * 4;                    // 32 K per step, 128 K per weight matrix
    for (int kb = 0; kb < kbMax; ++kb) {
        const int w    = kb >> 2;
        const int koff = (kb & 3) * 32;

        // --- A fragment: 16x32 bf16, lane = row (lo half K 0..7,16..23; hi half K 8..15,24..31)
        const bf16_t* ap = Sb + ((size_t)w * (size_t)N + (size_t)arow) * FD + koff + half * 8;
        union { v16bf v; uint4 q[2]; } A;
        A.q[0] = *(const uint4*)(ap);
        A.q[1] = *(const uint4*)(ap + 16);

        // --- B fragments: 32x16 bf16, lane = col (lo half K 0..7,16..23; hi half K 8..15,24..31)
        const bf16_t* bp = Wb + ((size_t)w * FD + (size_t)c) * FD + koff + half * 8;
#pragma unroll
        for (int n = 0; n < 8; ++n) {
            union { v16bf v; uint4 q[2]; } B;
            const bf16_t* bpn = bp + (size_t)n * 16 * FD;
            B.q[0] = *(const uint4*)(bpn);
            B.q[1] = *(const uint4*)(bpn + 16);
            acc[n] = __builtin_amdgcn_wmma_f32_16x16x32_bf16(
                false, A.v, false, B.v, (short)0, acc[n], false, false);
        }
    }

    // --- store C with fused tanh. C layout: VGPR r -> M = r + 8*half, N = lane&15
#pragma unroll
    for (int n = 0; n < 8; ++n) {
        const int col = n * 16 + c;
#pragma unroll
        for (int r = 0; r < 8; ++r) {
            const long row = rowBase + r + half * 8;
            if (row < N) out[row * FD + col] = tanhf(acc[n][r]);
        }
    }
}

// ---------- launcher ----------
extern "C" void kernel_launch(void* const* d_in, const int* in_sizes, int n_in,
                              void* d_out, int out_size, void* d_ws, size_t ws_size,
                              hipStream_t stream) {
    const float* x    = (const float*)d_in[0];
    const float* W    = (const float*)d_in[1];
    const int*   u    = (const int*)d_in[2];
    const int*   v    = (const int*)d_in[3];
    const int*   widx = (const int*)d_in[4];
    float*       out  = (float*)d_out;

    const int  N  = in_sizes[0] / FD;
    const int  NW = in_sizes[1] / (FD * FD);
    const long E  = in_sizes[2];

    // workspace layout: [ S_bf16 : NW*N*128 bf16 ][ W_bf16 : NW*128*128 bf16 ]
    bf16_t* Sb = (bf16_t*)d_ws;
    const size_t Sbytes = (size_t)NW * (size_t)N * FD * sizeof(bf16_t);
    bf16_t* Wb = (bf16_t*)((char*)d_ws + Sbytes);

    // 1) zero S (every call -> deterministic under graph replay)
    const long n16 = (long)(Sbytes / 16);
    zero_kernel<<<dim3((unsigned)((n16 + 255) / 256)), dim3(256), 0, stream>>>((uint4*)Sb, n16);

    // 2) W fp32 -> bf16
    const long npairs = (long)NW * FD * FD / 2;
    convw_kernel<<<dim3((unsigned)((npairs + 255) / 256)), dim3(256), 0, stream>>>(
        W, (unsigned*)Wb, npairs);

    // 3) edge scatter (1 wave per edge, 8 edges per block)
    scatter_kernel<<<dim3((unsigned)((E + 7) / 8)), dim3(256), 0, stream>>>(
        x, u, v, widx, Sb, N, E);

    // 4) fused grouped GEMM + tanh
    const unsigned mblocks = (unsigned)((N + 127) / 128);
    gemm_tanh_kernel<<<dim3(mblocks), dim3(256), 0, stream>>>(Sb, Wb, out, N, NW);
}
